// FrequencyGuidedSSDBlock_20993800143236
// MI455X (gfx1250) — matrix operand order repaired and checked by reading, not compiled, not run on hardware
//
#include <hip/hip_runtime.h>
#include <hip/hip_bf16.h>
#include <math.h>
#include <stdint.h>

// ---------------------------------------------------------------------------
// Types / helpers
// ---------------------------------------------------------------------------
typedef __attribute__((ext_vector_type(16))) __bf16 v16bf;
typedef __attribute__((ext_vector_type(8)))  float  v8f;

__device__ __forceinline__ float sigmoidf_(float x) { return 1.0f / (1.0f + expf(-x)); }
__device__ __forceinline__ float gelu_(float x) {
    return 0.5f * x * (1.0f + erff(x * 0.70710678118654752440f));   // exact GELU
}
__device__ __forceinline__ void bf16split(float f, __bf16& h, __bf16& l) {
    h = (__bf16)f;
    l = (__bf16)(f - (float)h);
}

// gfx1250 async copy global->LDS (ASYNCcnt path), no builtin on ROCm 7.2 ->
// inline asm per CDNA5 ISA (GV addressing mode).
__device__ __forceinline__ void async_copy_b128(uint32_t lds_off, const void* gaddr) {
    asm volatile("global_load_async_to_lds_b128 %0, %1, off"
                 :: "v"(lds_off), "v"(gaddr) : "memory");
}
__device__ __forceinline__ void wait_async_le6() {
    asm volatile("s_wait_asynccnt 0x6" ::: "memory");
}

// ---------------------------------------------------------------------------
// LayerNorm: one row (D=1024) per 256-thread block; emits f32 + bf16 hi/lo
// ---------------------------------------------------------------------------
__global__ __launch_bounds__(256) void ln_kernel(
    const float* __restrict__ x, const float* __restrict__ g, const float* __restrict__ b,
    float* __restrict__ outF, __bf16* __restrict__ outH, __bf16* __restrict__ outL, int D)
{
    const int row = blockIdx.x;
    const int tid = threadIdx.x;
    const float* xr = x + (size_t)row * D;

    float s = 0.0f, ss = 0.0f;
    for (int i = tid; i < D; i += 256) { float v = xr[i]; s += v; ss += v * v; }
    #pragma unroll
    for (int o = 16; o > 0; o >>= 1) {
        s  += __shfl_xor(s,  o, 32);
        ss += __shfl_xor(ss, o, 32);
    }
    __shared__ float rs[8], rss[8];
    if ((tid & 31) == 0) { rs[tid >> 5] = s; rss[tid >> 5] = ss; }
    __syncthreads();
    s = 0.0f; ss = 0.0f;
    #pragma unroll
    for (int i = 0; i < 8; i++) { s += rs[i]; ss += rss[i]; }

    const float mean = s / (float)D;
    const float var  = ss / (float)D - mean * mean;     // biased var (jnp.var)
    const float inv  = rsqrtf(var + 1e-5f);

    const size_t ro = (size_t)row * D;
    for (int i = tid; i < D; i += 256) {
        float v = (xr[i] - mean) * inv * g[i] + b[i];
        outF[ro + i] = v;
        __bf16 h, l; bf16split(v, h, l);
        outH[ro + i] = h; outL[ro + i] = l;
    }
}

// ---------------------------------------------------------------------------
// One-time weight prep: W[K,N] f32 -> transposed hi/lo bf16 planes [N,K].
// 64x64 LDS tile transpose, coalesced both directions.
// ---------------------------------------------------------------------------
__global__ __launch_bounds__(256) void wsplit_kernel(
    const float* __restrict__ W, int K, int N,
    __bf16* __restrict__ WhT, __bf16* __restrict__ WlT)
{
    __shared__ float tile[64][65];
    const int bk = blockIdx.y * 64, bn = blockIdx.x * 64;
    const int tid = threadIdx.x;
    const int tr  = tid >> 4;     // 0..15
    const int tc4 = tid & 15;     // float4 column group

    #pragma unroll
    for (int s = 0; s < 4; s++) {
        int r = tr + s * 16;
        float4 v = *reinterpret_cast<const float4*>(&W[(size_t)(bk + r) * N + bn + tc4 * 4]);
        tile[r][tc4 * 4 + 0] = v.x; tile[r][tc4 * 4 + 1] = v.y;
        tile[r][tc4 * 4 + 2] = v.z; tile[r][tc4 * 4 + 3] = v.w;
    }
    __syncthreads();
    #pragma unroll
    for (int s = 0; s < 4; s++) {
        int n = tr + s * 16;
        #pragma unroll
        for (int e = 0; e < 4; e++) {
            int k = tc4 * 4 + e;
            float f = tile[k][n];
            __bf16 h, l; bf16split(f, h, l);
            size_t o = (size_t)(bn + n) * K + bk + k;
            WhT[o] = h; WlT[o] = l;
        }
    }
}

// ---------------------------------------------------------------------------
// GEMM: C[M,N] = epi(A @ W + bias).  All operands pre-split bf16 hi/lo.
// Split-precision (3x v_wmma_f32_16x16x32_bf16 per tile) ~ fp32 accuracy.
// BM=128, BN=64, BK=32; 8 waves, each owns a 32x32 patch (2x2 WMMA tiles).
// Tiles staged in LDS with double-buffered GLOBAL_LOAD_ASYNC_TO_LDS_B128
// (6 async b128 per thread per stage; s_wait_asynccnt 0x6 retires a stage).
// Split-K dual A pointer feeds the [ssd|mamba] concat GEMM without concat.
// ---------------------------------------------------------------------------
#define BM 128
#define BN 64
#define BK 32
#define AROW 40                        // bf16 elems per LDS row (32 + 8 pad)
#define A_PLANE_B (BM * AROW * 2)      // 10240 B
#define B_PLANE_B (BN * AROW * 2)      // 5120 B
#define A_BUF_B   (2 * A_PLANE_B)      // hi+lo
#define B_BUF_B   (2 * B_PLANE_B)
#define SMEM_B    (2 * A_BUF_B + 2 * B_BUF_B)   // 61440 B

enum { EPI_NONE = 0, EPI_FR = 1, EPI_GELU = 2, EPI_SIGGATE = 3, EPI_ADD = 4 };

template <int EPI, bool WF32, bool WBF>
__global__ __launch_bounds__(256) void gemm_async(
    const __bf16* __restrict__ Ah,  const __bf16* __restrict__ Al,
    const __bf16* __restrict__ A2h, const __bf16* __restrict__ A2l,
    int K, int splitK,
    const __bf16* __restrict__ WhT, const __bf16* __restrict__ WlT,
    const float* __restrict__ bias, int N,
    const float* __restrict__ aux0, const float* __restrict__ aux1,
    const float* __restrict__ aux2,
    float* __restrict__ outF, __bf16* __restrict__ outH, __bf16* __restrict__ outL,
    __bf16* __restrict__ out1H, __bf16* __restrict__ out1L)
{
    __shared__ __align__(16) unsigned char smem[SMEM_B];
    const uint32_t smem0 = (uint32_t)(uintptr_t)(&smem[0]);

    const int tid  = threadIdx.x;
    const int lane = tid & 31;
    const int wave = tid >> 5;
    const int wm   = wave >> 1;          // 0..3 : 32-row band
    const int wn   = wave & 1;           // 0..1 : 32-col band
    const int m0   = blockIdx.y * BM;
    const int n0   = blockIdx.x * BN;
    const int K2   = K - splitK;

    // issue one stage: A tile 128x32 hi/lo + B tile 64x32 hi/lo (bf16)
    auto issue = [&](int buf, int k0) {
        #pragma unroll
        for (int s = 0; s < 2; s++) {
            int slot = tid + s * 256;               // 0..511 16B chunks of A
            int r    = slot >> 2;                   // row 0..127
            int cq   = slot & 3;                    // 16B chunk in row
            int gk   = k0 + cq * 8;
            int gm   = m0 + r;
            uint32_t loff = smem0 + (uint32_t)(buf * A_BUF_B + r * (AROW * 2) + cq * 16);
            const __bf16 *ph, *pl; size_t gi;
            if (gk < splitK) { ph = Ah;  pl = Al;  gi = (size_t)gm * splitK + gk; }
            else             { ph = A2h; pl = A2l; gi = (size_t)gm * K2 + (gk - splitK); }
            async_copy_b128(loff,             ph + gi);
            async_copy_b128(loff + A_PLANE_B, pl + gi);
        }
        {
            int r  = tid >> 2;                      // 0..63
            int cq = tid & 3;
            int gk = k0 + cq * 8;
            size_t gi = (size_t)(n0 + r) * K + gk;  // WT is [N][K]
            uint32_t loff = smem0 + (uint32_t)(2 * A_BUF_B + buf * B_BUF_B +
                                               r * (AROW * 2) + cq * 16);
            async_copy_b128(loff,             WhT + gi);
            async_copy_b128(loff + B_PLANE_B, WlT + gi);
        }
    };

    v8f acc[2][2];
    #pragma unroll
    for (int i = 0; i < 2; i++)
        #pragma unroll
        for (int j = 0; j < 2; j++)
            #pragma unroll
            for (int e = 0; e < 8; e++) acc[i][j][e] = 0.0f;

    issue(0, 0);                                    // preload buffer 0

    const int KT = K / BK;
    for (int kt = 0; kt < KT; kt++) {
        const int buf = kt & 1;
        __syncthreads();                 // all waves done reading buf^1 (previous iter)
        const int knext = (kt + 1 < KT) ? (kt + 1) * BK : 0;   // tail: harmless dummy
        issue(buf ^ 1, knext);
        wait_async_le6();                // retire current buffer's 6 copies
        __syncthreads();                 // current buffer visible to all waves

        const __bf16* pAh = (const __bf16*)(smem + buf * A_BUF_B);
        const __bf16* pAl = (const __bf16*)(smem + buf * A_BUF_B + A_PLANE_B);
        const __bf16* pBh = (const __bf16*)(smem + 2 * A_BUF_B + buf * B_BUF_B);
        const __bf16* pBl = (const __bf16*)(smem + 2 * A_BUF_B + buf * B_BUF_B + B_PLANE_B);

        // fragments per ISA 7.12.2 bf16 layouts
        v16bf aHi[2], aLo[2], bHi[2], bLo[2];
        const int mr  = lane & 15;
        const int kbA = (lane >> 4) << 3;    // 0 or 8
        #pragma unroll
        for (int mi = 0; mi < 2; mi++) {
            int row = wm * 32 + mi * 16 + mr;
            #pragma unroll
            for (int e = 0; e < 16; e++) {
                int k = ((e >> 3) << 4) + kbA + (e & 7);
                aHi[mi][e] = pAh[row * AROW + k];
                aLo[mi][e] = pAl[row * AROW + k];
            }
        }
        const int kbB = (lane >> 4) << 3;    // 0 or 8
        #pragma unroll
        for (int ni = 0; ni < 2; ni++) {
            int col = wn * 32 + ni * 16 + mr;
            #pragma unroll
            for (int e = 0; e < 16; e++) {
                bHi[ni][e] = pBh[col * AROW + ((e >> 3) << 4) + kbB + (e & 7)];
                bLo[ni][e] = pBl[col * AROW + ((e >> 3) << 4) + kbB + (e & 7)];
            }
        }

        #pragma unroll
        for (int mi = 0; mi < 2; mi++)
            #pragma unroll
            for (int ni = 0; ni < 2; ni++) {
                acc[mi][ni] = __builtin_amdgcn_wmma_f32_16x16x32_bf16(
                    false, aHi[mi], false, bHi[ni], (short)0, acc[mi][ni], false, false);
                acc[mi][ni] = __builtin_amdgcn_wmma_f32_16x16x32_bf16(
                    false, aHi[mi], false, bLo[ni], (short)0, acc[mi][ni], false, false);
                acc[mi][ni] = __builtin_amdgcn_wmma_f32_16x16x32_bf16(
                    false, aLo[mi], false, bHi[ni], (short)0, acc[mi][ni], false, false);
            }
    }

    // epilogue: C 16x16 f32 layout -> M = r + 8*(lane>=16), N = lane&15
    const int mhalf = (lane >> 4) * 8;
    const int ncol  = lane & 15;
    #pragma unroll
    for (int mi = 0; mi < 2; mi++)
        #pragma unroll
        for (int ni = 0; ni < 2; ni++) {
            const int   gn = n0 + wn * 32 + ni * 16 + ncol;
            const float bv = bias[gn];
            #pragma unroll
            for (int r = 0; r < 8; r++) {
                const int    gm  = m0 + wm * 32 + mi * 16 + mhalf + r;
                const size_t idx = (size_t)gm * N + gn;
                float v = acc[mi][ni][r] + bv;
                if (EPI == EPI_FR) {
                    float s    = sigmoidf_(v);
                    float xnv  = aux0[idx];
                    float lowv = xnv * s;
                    float highv = xnv - lowv;
                    __bf16 h, l;
                    bf16split(lowv,  h, l); outH[idx]  = h; outL[idx]  = l;
                    bf16split(highv, h, l); out1H[idx] = h; out1L[idx] = l;
                } else {
                    float res;
                    if (EPI == EPI_GELU)         res = gelu_(v);
                    else if (EPI == EPI_SIGGATE) {
                        float g = sigmoidf_(v);
                        res = aux0[idx] + g * aux1[idx] + (1.0f - g) * aux2[idx];
                    } else if (EPI == EPI_ADD)   res = aux0[idx] + v;
                    else                         res = v;
                    if (WF32) outF[idx] = res;
                    if (WBF) { __bf16 h, l; bf16split(res, h, l); outH[idx] = h; outL[idx] = l; }
                }
            }
        }
}

// ---------------------------------------------------------------------------
// Chunked parallel scan (affine recurrence is associative).
// channels = B*D = 4096, L = 2048 in 16 chunks of 128.
// ---------------------------------------------------------------------------
#define SC_B 4
#define SC_L 2048
#define SC_D 1024
#define SC_NCH (SC_B * SC_D)
#define SC_NCHUNK 16
#define SC_CL (SC_L / SC_NCHUNK)

__global__ __launch_bounds__(256) void scan_phase1(
    const float* __restrict__ delta, const float* __restrict__ bterm,
    const float* __restrict__ xn, float* __restrict__ Ac, float* __restrict__ Bc)
{
    const int g     = blockIdx.x * 256 + threadIdx.x;
    const int c     = g & (SC_NCH - 1);
    const int chunk = g >> 12;
    const int b     = c >> 10, d = c & (SC_D - 1);
    float A = 1.0f, Bv = 0.0f;
    size_t base = ((size_t)b * SC_L + (size_t)chunk * SC_CL) * SC_D + d;
    for (int l = 0; l < SC_CL; l++) {
        size_t idx = base + (size_t)l * SC_D;
        float a  = sigmoidf_(delta[idx]);
        float bb = tanhf(bterm[idx]) * xn[idx];
        A  = A * a;
        Bv = a * Bv + bb;
    }
    Ac[g] = A; Bc[g] = Bv;
}

__global__ __launch_bounds__(256) void scan_phase2(
    const float* __restrict__ Ac, const float* __restrict__ Bc, float* __restrict__ init)
{
    const int c = blockIdx.x * 256 + threadIdx.x;
    float s = 0.0f;
    #pragma unroll
    for (int ch = 0; ch < SC_NCHUNK; ch++) {
        int g = ch * SC_NCH + c;
        init[g] = s;
        s = Ac[g] * s + Bc[g];
    }
}

__global__ __launch_bounds__(256) void scan_phase3(
    const float* __restrict__ delta, const float* __restrict__ bterm,
    const float* __restrict__ cterm, const float* __restrict__ xn,
    const float* __restrict__ skip, const float* __restrict__ init,
    float* __restrict__ out, __bf16* __restrict__ outH, __bf16* __restrict__ outL)
{
    const int g     = blockIdx.x * 256 + threadIdx.x;
    const int c     = g & (SC_NCH - 1);
    const int chunk = g >> 12;
    const int b     = c >> 10, d = c & (SC_D - 1);
    float state = init[g];
    const float sk = skip[d];
    size_t base = ((size_t)b * SC_L + (size_t)chunk * SC_CL) * SC_D + d;
    for (int l = 0; l < SC_CL; l++) {
        size_t idx = base + (size_t)l * SC_D;
        float xv = xn[idx];
        float a  = sigmoidf_(delta[idx]);
        state = a * state + tanhf(bterm[idx]) * xv;
        float r = tanhf(cterm[idx]) * state + sk * xv;
        out[idx] = r;
        __bf16 h, l2; bf16split(r, h, l2);
        outH[idx] = h; outL[idx] = l2;
    }
}

// ---------------------------------------------------------------------------
// Launcher
// ---------------------------------------------------------------------------
extern "C" void kernel_launch(void* const* d_in, const int* in_sizes, int n_in,
                              void* d_out, int out_size, void* d_ws, size_t ws_size,
                              hipStream_t stream)
{
    const float* x      = (const float*)d_in[0];
    const float* norm_g = (const float*)d_in[1];
    const float* norm_b = (const float*)d_in[2];
    const float* fr_w   = (const float*)d_in[3];
    const float* fr_b   = (const float*)d_in[4];
    const float* dp_w   = (const float*)d_in[5];
    const float* dp_b   = (const float*)d_in[6];
    const float* bp_w   = (const float*)d_in[7];
    const float* bp_b   = (const float*)d_in[8];
    const float* cp_w   = (const float*)d_in[9];
    const float* cp_b   = (const float*)d_in[10];
    const float* skip   = (const float*)d_in[11];
    const float* m1_w   = (const float*)d_in[12];
    const float* m1_b   = (const float*)d_in[13];
    const float* m2_w   = (const float*)d_in[14];
    const float* m2_b   = (const float*)d_in[15];
    const float* gate_w = (const float*)d_in[16];
    const float* gate_b = (const float*)d_in[17];
    const float* ffn_g  = (const float*)d_in[18];
    const float* ffn_bn = (const float*)d_in[19];
    const float* ffn1_w = (const float*)d_in[20];
    const float* ffn1_b = (const float*)d_in[21];
    const float* ffn2_w = (const float*)d_in[22];
    const float* ffn2_b = (const float*)d_in[23];
    float* out = (float*)d_out;

    const int    M  = 8192, D = 1024;
    const size_t MD = (size_t)M * D;

    // --- workspace carve ---------------------------------------------------
    char* p = (char*)d_ws;
    auto carveF = [&](size_t n) { float*  q = (float*)p;  p += n * sizeof(float);  return q; };
    auto carveB = [&](size_t n) { __bf16* q = (__bf16*)p; p += n * sizeof(__bf16); return q; };

    float* xnF    = carveF(MD);
    float* yF     = carveF(MD);
    float* ynF    = carveF(MD);
    float* mambaF = carveF(MD);
    // contiguous 4*MD f32 region, dead after gate GEMM -> reused for ffn hidden
    float* deltaF = carveF(MD);
    float* btermF = carveF(MD);
    float* ctermF = carveF(MD);
    float* ssdF   = carveF(MD);
    __bf16* ffnhH = (__bf16*)deltaF;          // alias: 4*MD bf16
    __bf16* ffnhL = ffnhH + 4 * MD;           // alias: next 4*MD bf16 (ends at ssdF end)

    __bf16* xnH   = carveB(MD);  __bf16* xnL   = carveB(MD);
    __bf16* lowH  = carveB(MD);  __bf16* lowL  = carveB(MD);
    __bf16* highH = carveB(MD);  __bf16* highL = carveB(MD);
    __bf16* ssdH  = carveB(MD);  __bf16* ssdL  = carveB(MD);
    __bf16* mamH  = carveB(MD);  __bf16* mamL  = carveB(MD);
    __bf16* ynH   = carveB(MD);  __bf16* ynL   = carveB(MD);
    __bf16* h1H   = carveB(2 * MD); __bf16* h1L = carveB(2 * MD);

    const size_t WD = (size_t)D * D;          // 1M weight elements
    __bf16* frT[2]   = { carveB(WD),     carveB(WD)     };
    __bf16* dpT[2]   = { carveB(WD),     carveB(WD)     };
    __bf16* bpT[2]   = { carveB(WD),     carveB(WD)     };
    __bf16* cpT[2]   = { carveB(WD),     carveB(WD)     };
    __bf16* m1T[2]   = { carveB(2 * WD), carveB(2 * WD) };
    __bf16* m2T[2]   = { carveB(2 * WD), carveB(2 * WD) };
    __bf16* gateT[2] = { carveB(2 * WD), carveB(2 * WD) };
    __bf16* f1T[2]   = { carveB(4 * WD), carveB(4 * WD) };
    __bf16* f2T[2]   = { carveB(4 * WD), carveB(4 * WD) };

    float* scanA = carveF(SC_NCH * SC_NCHUNK);
    float* scanB = carveF(SC_NCH * SC_NCHUNK);
    float* scanI = carveF(SC_NCH * SC_NCHUNK);

    const dim3 blk(256);
    const dim3 gD (D / BN,       M / BM);
    const dim3 g2D((2 * D) / BN, M / BM);
    const dim3 g4D((4 * D) / BN, M / BM);

    // --- 0. split+transpose all weights to bf16 hi/lo [N][K] ---------------
    wsplit_kernel<<<dim3(D / 64,     D / 64),     blk, 0, stream>>>(fr_w,   D,     D,     frT[0],   frT[1]);
    wsplit_kernel<<<dim3(D / 64,     D / 64),     blk, 0, stream>>>(dp_w,   D,     D,     dpT[0],   dpT[1]);
    wsplit_kernel<<<dim3(D / 64,     D / 64),     blk, 0, stream>>>(bp_w,   D,     D,     bpT[0],   bpT[1]);
    wsplit_kernel<<<dim3(D / 64,     D / 64),     blk, 0, stream>>>(cp_w,   D,     D,     cpT[0],   cpT[1]);
    wsplit_kernel<<<dim3(2 * D / 64, D / 64),     blk, 0, stream>>>(m1_w,   D,     2 * D, m1T[0],   m1T[1]);
    wsplit_kernel<<<dim3(D / 64,     2 * D / 64), blk, 0, stream>>>(m2_w,   2 * D, D,     m2T[0],   m2T[1]);
    wsplit_kernel<<<dim3(D / 64,     2 * D / 64), blk, 0, stream>>>(gate_w, 2 * D, D,     gateT[0], gateT[1]);
    wsplit_kernel<<<dim3(4 * D / 64, D / 64),     blk, 0, stream>>>(ffn1_w, D,     4 * D, f1T[0],   f1T[1]);
    wsplit_kernel<<<dim3(D / 64,     4 * D / 64), blk, 0, stream>>>(ffn2_w, 4 * D, D,     f2T[0],   f2T[1]);

    // --- 1. xn = LN(x) ------------------------------------------------------
    ln_kernel<<<M, blk, 0, stream>>>(x, norm_g, norm_b, xnF, xnH, xnL, D);

    // --- 2. frequency gate -> low/high bf16 planes (fused) ------------------
    gemm_async<EPI_FR, false, true><<<gD, blk, 0, stream>>>(
        xnH, xnL, xnH, xnL, D, D, frT[0], frT[1], fr_b, D,
        xnF, nullptr, nullptr, nullptr, lowH, lowL, highH, highL);

    // --- 3-5. delta/bterm/cterm projections (f32 for the scan) --------------
    gemm_async<EPI_NONE, true, false><<<gD, blk, 0, stream>>>(
        lowH, lowL, lowH, lowL, D, D, dpT[0], dpT[1], dp_b, D,
        nullptr, nullptr, nullptr, deltaF, nullptr, nullptr, nullptr, nullptr);
    gemm_async<EPI_NONE, true, false><<<gD, blk, 0, stream>>>(
        lowH, lowL, lowH, lowL, D, D, bpT[0], bpT[1], bp_b, D,
        nullptr, nullptr, nullptr, btermF, nullptr, nullptr, nullptr, nullptr);
    gemm_async<EPI_NONE, true, false><<<gD, blk, 0, stream>>>(
        highH, highL, highH, highL, D, D, cpT[0], cpT[1], cp_b, D,
        nullptr, nullptr, nullptr, ctermF, nullptr, nullptr, nullptr, nullptr);

    // --- 6. chunked parallel selective scan -> ssd (f32 + planes) -----------
    scan_phase1<<<(SC_NCH * SC_NCHUNK) / 256, blk, 0, stream>>>(deltaF, btermF, xnF, scanA, scanB);
    scan_phase2<<<SC_NCH / 256, blk, 0, stream>>>(scanA, scanB, scanI);
    scan_phase3<<<(SC_NCH * SC_NCHUNK) / 256, blk, 0, stream>>>(
        deltaF, btermF, ctermF, xnF, skip, scanI, ssdF, ssdH, ssdL);

    // --- 7-8. mamba branch ---------------------------------------------------
    gemm_async<EPI_GELU, false, true><<<g2D, blk, 0, stream>>>(
        lowH, lowL, lowH, lowL, D, D, m1T[0], m1T[1], m1_b, 2 * D,
        nullptr, nullptr, nullptr, nullptr, h1H, h1L, nullptr, nullptr);
    gemm_async<EPI_NONE, true, true><<<gD, blk, 0, stream>>>(
        h1H, h1L, h1H, h1L, 2 * D, 2 * D, m2T[0], m2T[1], m2_b, D,
        nullptr, nullptr, nullptr, mambaF, mamH, mamL, nullptr, nullptr);

    // --- 9. gate over [ssd|mamba] via split-K, fused mix + residual -> y ----
    gemm_async<EPI_SIGGATE, true, false><<<gD, blk, 0, stream>>>(
        ssdH, ssdL, mamH, mamL, 2 * D, D, gateT[0], gateT[1], gate_b, D,
        x, ssdF, mambaF, yF, nullptr, nullptr, nullptr, nullptr);

    // --- 10. yn = LN(y) ------------------------------------------------------
    ln_kernel<<<M, blk, 0, stream>>>(yF, ffn_g, ffn_bn, ynF, ynH, ynL, D);

    // --- 11-12. FFN ----------------------------------------------------------
    gemm_async<EPI_GELU, false, true><<<g4D, blk, 0, stream>>>(
        ynH, ynL, ynH, ynL, D, D, f1T[0], f1T[1], ffn1_b, 4 * D,
        nullptr, nullptr, nullptr, nullptr, ffnhH, ffnhL, nullptr, nullptr);
    gemm_async<EPI_ADD, true, false><<<gD, blk, 0, stream>>>(
        ffnhH, ffnhL, ffnhH, ffnhL, 4 * D, 4 * D, f2T[0], f2T[1], ffn2_b, D,
        yF, nullptr, nullptr, out, nullptr, nullptr, nullptr, nullptr);

    (void)in_sizes; (void)n_in; (void)out_size; (void)ws_size;
}
